// PeerSelfAttention_9706626089811
// MI455X (gfx1250) — compile-verified, matrix operand order checked by reading, not run on hardware
//
#include <hip/hip_runtime.h>
#include <hip/hip_bf16.h>

typedef __bf16 bf16_t;
typedef __attribute__((ext_vector_type(16))) __bf16 v16bf;
typedef __attribute__((ext_vector_type(8)))  float  v8f;
typedef __attribute__((ext_vector_type(4)))  int    v4i;

#define WMMA_BF16(a, b, c) \
  __builtin_amdgcn_wmma_f32_16x16x32_bf16(false, (a), false, (b), (short)0, (c), false, false)

// Problem constants
#define BB 2
#define SS 2048
#define HH 1024
#define NHH 16
#define DHH 64

// Workspace byte offsets
#define OFF_HX 0ull
#define OFF_WQ 8388608ull
#define OFF_WK 10485760ull
#define OFF_WV 12582912ull
#define OFF_Q  14680064ull
#define OFF_K  23068672ull
#define OFF_V  31457280ull

// ---------------------------------------------------------------------------
// Async global->LDS copy (CDNA5 GLOBAL_LOAD_ASYNC_TO_LDS_B128, ASYNCcnt-
// tracked) with synchronous fallback if the builtin is unavailable.
// Builtin signature (from hipcc diagnostic): (v4i AS1*, v4i AS3*, Ii, Ii).
// ---------------------------------------------------------------------------
#if defined(__has_builtin)
#if __has_builtin(__builtin_amdgcn_global_load_async_to_lds_b128)
#define HAVE_ASYNC_LDS 1
#endif
#endif

__device__ __forceinline__ void cp128(bf16_t* lds, const bf16_t* g) {
#if defined(HAVE_ASYNC_LDS)
  __builtin_amdgcn_global_load_async_to_lds_b128(
      (__attribute__((address_space(1))) v4i*)(uintptr_t)(g),
      (__attribute__((address_space(3))) v4i*)(uintptr_t)(lds), 0, 0);
#else
  *(uint4*)lds = *(const uint4*)g;
#endif
}

__device__ __forceinline__ void wait_async0() {
#if defined(HAVE_ASYNC_LDS)
#if __has_builtin(__builtin_amdgcn_s_wait_asynccnt)
  __builtin_amdgcn_s_wait_asynccnt(0);
#else
  asm volatile("s_wait_asynccnt 0x0" ::: "memory");
#endif
#endif
}

__device__ __forceinline__ v16bf ld_frag(const bf16_t* p) {
  union { uint4 u[2]; v16bf v; } f;
  f.u[0] = *(const uint4*)(p);
  f.u[1] = *(const uint4*)(p + 16);
  return f.v;
}

// ---------------------------------------------------------------------------
// Kernel 0: fp32 -> bf16 conversion
// ---------------------------------------------------------------------------
__global__ void f32_to_bf16_kernel(const float* __restrict__ in,
                                   bf16_t* __restrict__ out, int n) {
  int i = blockIdx.x * blockDim.x + threadIdx.x;
  if (i < n) out[i] = (bf16_t)in[i];
}

// ---------------------------------------------------------------------------
// Kernel 1: fused QKV projection GEMM (double-buffered async LDS staging).
// out[m,n] = sum_k X[m,k] * W[n,k] + bias[n]   (bf16 in, f32 acc, bf16 out)
// grid = (N/64, M/128, 3), block = 256 (8 waves), wave grid 4(M) x 2(N),
// each wave computes 32x32 via 2x2 WMMA 16x16x32 frags.
// z==2 (V) stores transposed layout [B,NH,DH,S]; else [B,NH,S,DH].
// ---------------------------------------------------------------------------
__global__ __launch_bounds__(256) void qkv_gemm_kernel(
    const bf16_t* __restrict__ X,
    const bf16_t* __restrict__ Wqb, const bf16_t* __restrict__ Wkb,
    const bf16_t* __restrict__ Wvb,
    const float* __restrict__ bq, const float* __restrict__ bk,
    const float* __restrict__ bv,
    bf16_t* __restrict__ Qo, bf16_t* __restrict__ Ko, bf16_t* __restrict__ Vo) {
  const int zz = blockIdx.z;
  const bf16_t* W    = (zz == 0) ? Wqb : (zz == 1) ? Wkb : Wvb;
  const float*  bias = (zz == 0) ? bq  : (zz == 1) ? bk  : bv;
  bf16_t*       out  = (zz == 0) ? Qo  : (zz == 1) ? Ko  : Vo;
  const bool vmode = (zz == 2);

  const int n0 = blockIdx.x * 64;
  const int m0 = blockIdx.y * 128;

  __shared__ __align__(16) bf16_t sA[2][128 * 32];
  __shared__ __align__(16) bf16_t sW[2][64 * 32];

  const int t    = threadIdx.x;
  const int lane = t & 31;
  const int wave = t >> 5;
  const int wm = wave >> 1;      // 0..3
  const int wn = wave & 1;       // 0..1
  const int la   = lane & 15;
  const int koff = (lane & 16) ? 8 : 0;

  auto stage = [&](int buf, int k0) {
    // A tile 128x32 (512 x 16B, 2 per thread)
    for (int i = 0; i < 2; ++i) {
      int u = t + i * 256;
      int row = u >> 2, c = u & 3;
      cp128(sA[buf] + u * 8, X + (size_t)(m0 + row) * HH + k0 + c * 8);
    }
    // W tile 64x32 (256 x 16B, 1 per thread)
    {
      int row = t >> 2, c = t & 3;
      cp128(sW[buf] + t * 8, W + (size_t)(n0 + row) * HH + k0 + c * 8);
    }
  };

  v8f acc[2][2] = {};

  stage(0, 0);
  int cur = 0;
  for (int k0 = 0; k0 < HH; k0 += 32) {
    wait_async0();
    __syncthreads();
    if (k0 + 32 < HH) stage(cur ^ 1, k0 + 32);

    v16bf af[2], bfr[2];
    for (int mi = 0; mi < 2; ++mi)
      af[mi] = ld_frag(sA[cur] + (wm * 32 + mi * 16 + la) * 32 + koff);
    for (int ni = 0; ni < 2; ++ni)
      bfr[ni] = ld_frag(sW[cur] + (wn * 32 + ni * 16 + la) * 32 + koff);
    for (int mi = 0; mi < 2; ++mi)
      for (int ni = 0; ni < 2; ++ni)
        acc[mi][ni] = WMMA_BF16(af[mi], bfr[ni], acc[mi][ni]);
    cur ^= 1;
  }

  // Epilogue: add bias, cast to bf16, scatter into head layout.
  const int mrow0 = (lane & 16) ? 8 : 0;
  for (int mi = 0; mi < 2; ++mi) {
    for (int ni = 0; ni < 2; ++ni) {
      const int n_glob = n0 + wn * 32 + ni * 16 + la;
      const float bn = bias[n_glob];
      const int h = n_glob >> 6;
      const int d = n_glob & 63;
      for (int r = 0; r < 8; ++r) {
        const int m_glob = m0 + wm * 32 + mi * 16 + mrow0 + r;
        const int b = m_glob >> 11;       // /2048
        const int s = m_glob & 2047;
        const float v = acc[mi][ni][r] + bn;
        size_t idx;
        if (vmode)
          idx = (((size_t)(b * NHH + h)) * DHH + d) * SS + s;   // [B,NH,DH,S]
        else
          idx = (((size_t)(b * NHH + h)) * SS + s) * DHH + d;   // [B,NH,S,DH]
        out[idx] = (bf16_t)v;
      }
    }
  }
}

// ---------------------------------------------------------------------------
// Kernel 2: flash attention with relative scalar bias.
// grid = (S/64, NH, B), block = 128 (4 waves, each owns 16 q-rows).
// K/V tiles double-buffered through async global->LDS copies.
// ---------------------------------------------------------------------------
__global__ __launch_bounds__(128) void attn_kernel(
    const bf16_t* __restrict__ Qb, const bf16_t* __restrict__ Kb,
    const bf16_t* __restrict__ Vtb, const int* __restrict__ didx,
    const float* __restrict__ demb, const float* __restrict__ mask,
    float* __restrict__ out) {
  const int qt = blockIdx.x;
  const int h  = blockIdx.y;
  const int b  = blockIdx.z;
  const int q0 = qt * 64;

  __shared__ __align__(16) bf16_t sK[2][64 * 64];    // [k][d]
  __shared__ __align__(16) bf16_t sV[2][64 * 64];    // [d][k]  (pre-transposed V)
  __shared__ __align__(16) bf16_t sP[4][16 * 64];    // per-wave P staging

  const int t    = threadIdx.x;
  const int lane = t & 31;
  const int wave = t >> 5;
  const int la   = lane & 15;
  const int koff = (lane & 16) ? 8 : 0;

  const size_t bh = (size_t)(b * NHH + h);
  const bf16_t* Qp = Qb  + bh * SS * DHH;
  const bf16_t* Kp = Kb  + bh * SS * DHH;
  const bf16_t* Vp = Vtb + bh * DHH * SS;

  auto stageKV = [&](int buf, int kb) {
    // K tile [k][d]: contiguous 8KB (512 x 16B / 128 threads = 4 each)
    const bf16_t* gK = Kp + (size_t)kb * 64 * 64;
    for (int i = 0; i < 4; ++i) {
      int u = t + i * 128;
      cp128(sK[buf] + u * 8, gK + u * 8);
    }
    // Vt tile [d][k]: 64 rows of 128B from strided global rows
    for (int i = 0; i < 4; ++i) {
      int u = t + i * 128;
      int d = u >> 3, c = u & 7;
      cp128(sV[buf] + u * 8, Vp + (size_t)d * SS + kb * 64 + c * 8);
    }
  };

  // Q A-fragments for this wave's 16 rows (kept in registers throughout).
  const int qrow_a = q0 + wave * 16 + la;
  const bf16_t* qrp = Qp + (size_t)qrow_a * DHH;
  v16bf aQ[2];
  aQ[0] = ld_frag(qrp + 0  + koff);
  aQ[1] = ld_frag(qrp + 32 + koff);

  v8f O[4] = {};
  float m_i[8], l_i[8];
  for (int r = 0; r < 8; ++r) { m_i[r] = -3.0e38f; l_i[r] = 0.0f; }

  const int qrow_c0 = q0 + wave * 16 + ((lane & 16) ? 8 : 0);  // C-row base

  stageKV(0, 0);
  int cur = 0;
  for (int kb = 0; kb < SS / 64; ++kb) {
    wait_async0();
    __syncthreads();
    if (kb + 1 < SS / 64) stageKV(cur ^ 1, kb + 1);

    // scores = Q K^T for 16x64 strip (4 N-tiles x 2 K-steps)
    float s4[4][8];
    for (int j = 0; j < 4; ++j) {
      v8f acc = {};
      for (int ks = 0; ks < 2; ++ks) {
        v16bf bK = ld_frag(sK[cur] + (j * 16 + la) * 64 + ks * 32 + koff);
        acc = WMMA_BF16(aQ[ks], bK, acc);
      }
      for (int r = 0; r < 8; ++r) s4[j][r] = acc[r];
    }

    // scale + relative bias gather + mask (per C-fragment element)
    for (int j = 0; j < 4; ++j) {
      const int kcol = kb * 64 + j * 16 + la;
      const float mk = mask[b * SS + kcol];
      for (int r = 0; r < 8; ++r) {
        const int qg = qrow_c0 + r;
        const int di = didx[(size_t)qg * SS + kcol];
        s4[j][r] = s4[j][r] * 0.125f + demb[di * NHH + h] + mk;
      }
    }

    // online softmax (rows live in 16-lane halves; shfl_xor masks 1..8 stay in-half)
    float scale[8];
    for (int r = 0; r < 8; ++r) {
      float rm = fmaxf(fmaxf(s4[0][r], s4[1][r]), fmaxf(s4[2][r], s4[3][r]));
      for (int m = 1; m < 16; m <<= 1) rm = fmaxf(rm, __shfl_xor(rm, m, 32));
      const float mn = fmaxf(m_i[r], rm);
      scale[r] = __expf(m_i[r] - mn);
      m_i[r] = mn;
      float rs = 0.0f;
      for (int j = 0; j < 4; ++j) {
        const float p = __expf(s4[j][r] - mn);
        s4[j][r] = p;
        rs += p;
      }
      for (int m = 1; m < 16; m <<= 1) rs += __shfl_xor(rs, m, 32);
      l_i[r] = l_i[r] * scale[r] + rs;
    }
    for (int j = 0; j < 4; ++j)
      for (int r = 0; r < 8; ++r) O[j][r] *= scale[r];

    // Restage P (C-layout -> A-layout) through LDS as bf16
    bf16_t* pw = &sP[wave][0];
    const int mrow0 = (lane & 16) ? 8 : 0;
    for (int j = 0; j < 4; ++j)
      for (int r = 0; r < 8; ++r)
        pw[(mrow0 + r) * 64 + j * 16 + la] = (bf16_t)s4[j][r];
    asm volatile("s_wait_dscnt 0x0" ::: "memory");

    // O += P @ V  (contraction over 64 k's = 2 K-steps)
    for (int ks = 0; ks < 2; ++ks) {
      v16bf aP = ld_frag(pw + la * 64 + ks * 32 + koff);
      for (int j = 0; j < 4; ++j) {
        v16bf bV = ld_frag(sV[cur] + (j * 16 + la) * 64 + ks * 32 + koff);
        O[j] = WMMA_BF16(aP, bV, O[j]);
      }
    }
    cur ^= 1;
  }

  // Epilogue: O / l, write fp32 [B,S,H]
  float inv[8];
  for (int r = 0; r < 8; ++r) inv[r] = 1.0f / l_i[r];
  for (int j = 0; j < 4; ++j) {
    for (int r = 0; r < 8; ++r) {
      const int qg = qrow_c0 + r;
      out[((size_t)b * SS + qg) * HH + h * DHH + j * 16 + la] = O[j][r] * inv[r];
    }
  }
}

// ---------------------------------------------------------------------------
extern "C" void kernel_launch(void* const* d_in, const int* in_sizes, int n_in,
                              void* d_out, int out_size, void* d_ws, size_t ws_size,
                              hipStream_t stream) {
  const float* hidden = (const float*)d_in[0];
  const float* amask  = (const float*)d_in[1];
  const int*   didx   = (const int*)d_in[2];
  const float* Wq     = (const float*)d_in[3];
  const float* bq     = (const float*)d_in[4];
  const float* Wk     = (const float*)d_in[5];
  const float* bk     = (const float*)d_in[6];
  const float* Wv     = (const float*)d_in[7];
  const float* bv     = (const float*)d_in[8];
  const float* demb   = (const float*)d_in[9];
  float* out = (float*)d_out;

  char* ws = (char*)d_ws;
  bf16_t* hX = (bf16_t*)(ws + OFF_HX);
  bf16_t* wq = (bf16_t*)(ws + OFF_WQ);
  bf16_t* wk = (bf16_t*)(ws + OFF_WK);
  bf16_t* wv = (bf16_t*)(ws + OFF_WV);
  bf16_t* Qb = (bf16_t*)(ws + OFF_Q);
  bf16_t* Kb = (bf16_t*)(ws + OFF_K);
  bf16_t* Vb = (bf16_t*)(ws + OFF_V);

  const int nHid = BB * SS * HH;   // 4194304
  const int nW   = HH * HH;        // 1048576

  f32_to_bf16_kernel<<<nHid / 256, 256, 0, stream>>>(hidden, hX, nHid);
  f32_to_bf16_kernel<<<nW / 256, 256, 0, stream>>>(Wq, wq, nW);
  f32_to_bf16_kernel<<<nW / 256, 256, 0, stream>>>(Wk, wk, nW);
  f32_to_bf16_kernel<<<nW / 256, 256, 0, stream>>>(Wv, wv, nW);

  qkv_gemm_kernel<<<dim3(HH / 64, (BB * SS) / 128, 3), 256, 0, stream>>>(
      hX, wq, wk, wv, bq, bk, bv, Qb, Kb, Vb);

  attn_kernel<<<dim3(SS / 64, NHH, BB), 128, 0, stream>>>(
      Qb, Kb, Vb, didx, demb, amask, out);
}
